// TrendGenerative_42700564857075
// MI455X (gfx1250) — compile-verified, matrix operand order checked by reading, not run on hardware
//
#include <hip/hip_runtime.h>
#include <cmath>

typedef __attribute__((ext_vector_type(16))) _Float16 v16h;
typedef __attribute__((ext_vector_type(8)))  float    v8f;

#define B  128
#define T  512
#define M  64
#define DD 16      // time dim D
#define KP 8       // prototypes per group (K)
#define K2 16      // 2K

// workspace layout (bytes)
#define WS_E    0u                 // B*T*32 _Float16  = 4,194,304 B
#define WS_WH   4194304u           // 2K*M*2D _Float16 = 65,536 B
#define WS_BERN 4259840u           // B floats
#define WS_DIST 4260352u           // B*2K floats

__global__ void init_kernel(float* bern, float* dist) {
  int i = blockIdx.x * 256 + threadIdx.x;
  if (i < B) bern[i] = 0.f;
  if (i < B * K2) dist[i] = 0.f;
}

__global__ void wconv_kernel(const float* __restrict__ W, _Float16* __restrict__ Wh) {
  int i = blockIdx.x * 256 + threadIdx.x;      // 32768 total
  Wh[i] = (_Float16)W[i];
}

__global__ void eprep_kernel(const float* __restrict__ test_tl, const int* __restrict__ n_task,
                             const float* __restrict__ basis, const float* __restrict__ phase,
                             _Float16* __restrict__ E) {
  int tid = blockIdx.x * 256 + threadIdx.x;    // B*T*32 total
  int d  = tid & 31;
  int bt = tid >> 5;
  int t  = bt & (T - 1);
  int b  = bt >> 9;
  float x   = test_tl[bt];
  int   di  = d & 15;
  float arg = x * basis[di] + phase[di];
  float v   = (d < DD ? __cosf(arg) : __sinf(arg)) * 0.25f;  // scale = 1/sqrt(16)
  if (t >= n_task[b]) v = 0.f;                               // mask baked into Phi
  E[tid] = (_Float16)v;
}

__global__ void __launch_bounds__(256)
wmma_kernel(const float* __restrict__ sim, const _Float16* __restrict__ E,
            const _Float16* __restrict__ Wh, float* bern, float* dist) {
  __shared__ __align__(16) _Float16 wlds[KP * M * 2 * DD];   // 16384 halfs = 32 KB
  const int tid  = threadIdx.x;
  const int lane = tid & 31;
  const int wid  = tid >> 5;
  const int task = blockIdx.x * 8 + wid;   // (b, t_tile)
  const int b     = task >> 5;
  const int ttile = task & 31;
  const int t0    = ttile * 16;
  const int lo = lane & 15;
  const int hi = lane >> 4;

  // A-tile (16x32 f16): lane = row t0+lo, K-half selected by hi -> 32B contiguous
  v16h a = *(const v16h*)(E + ((size_t)(b * T + t0 + lo)) * 32 + hi * 16);

  // S = inv_sigmoid(sim) at the D-matrix positions; bern partial on the fly
  float sv[4][8];
  float bacc = 0.f;
#pragma unroll
  for (int mt = 0; mt < 4; ++mt) {
#pragma unroll
    for (int r = 0; r < 8; ++r) {
      int trow = t0 + r + 8 * hi;
      float x  = sim[((size_t)(b * T + trow)) * M + mt * 16 + lo];
      float l1 = __logf(x);
      float l2 = __logf(1.0f - x);
      float s  = l1 - l2;
      if (isinf(s)) s = (s < 0.f) ? 0.f : 3.402823466e38f;   // nan_to_num(neginf=0)
      if (isnan(s)) s = 0.f;
      sv[mt][r] = s;
      float bb = l1 + l2;                                    // log(x)+log(1-x)
      if (!isfinite(bb)) bb = 0.f;
      bacc += bb;
    }
  }
#pragma unroll
  for (int m = 1; m < 32; m <<= 1) bacc += __shfl_xor(bacc, m, 32);
  if (lane == 0) atomicAdd(&bern[b], bacc);

  // two phases of 8 prototypes; W staged in LDS
  for (int h = 0; h < 2; ++h) {
    __syncthreads();
    const uint4* src = (const uint4*)(Wh + h * (KP * M * 2 * DD));
    uint4*       dst = (uint4*)wlds;
#pragma unroll
    for (int i = 0; i < 8; ++i) dst[tid + i * 256] = src[tid + i * 256];
    __syncthreads();

    for (int kk = 0; kk < KP; ++kk) {
      float psum[8];
#pragma unroll
      for (int r = 0; r < 8; ++r) psum[r] = 0.f;
#pragma unroll
      for (int mt = 0; mt < 4; ++mt) {
        // B-tile (32x16 f16): lane = col m, K-half by hi -> contiguous d in W[k,m,:]
        v16h bm = *(const v16h*)(wlds + (kk * M + mt * 16 + lo) * 32 + hi * 16);
        v8f c = {};
        v8f dacc = __builtin_amdgcn_wmma_f32_16x16x32_f16(
            false, a, false, bm, (short)0, c, false, false);
#pragma unroll
        for (int r = 0; r < 8; ++r) {
          float diff = sv[mt][r] - dacc[r];
          psum[r] += diff * diff;
        }
      }
      // per-row L2 norm across the 16 lanes of each half, then sum rows
      float acc = 0.f;
#pragma unroll
      for (int r = 0; r < 8; ++r) {
        float rs = psum[r];
        rs += __shfl_xor(rs, 1, 32);
        rs += __shfl_xor(rs, 2, 32);
        rs += __shfl_xor(rs, 4, 32);
        rs += __shfl_xor(rs, 8, 32);
        acc += sqrtf(rs);
      }
      acc += __shfl_xor(acc, 16, 32);
      if (lane == 0) atomicAdd(&dist[b * K2 + h * KP + kk], acc);
    }
  }
}

__global__ void final_kernel(const int* __restrict__ n_task, const float* __restrict__ mlp_w,
                             const float* __restrict__ bern, const float* __restrict__ dist,
                             float* __restrict__ out) {
  int b = threadIdx.x;
  if (b >= B) return;
  float nt = (float)n_task[b];
  float bb = bern[b] / nt;              // BETA = 1.0
  float score0 = 0.f, score1 = 0.f;
#pragma unroll
  for (int k = 0; k < KP; ++k) {
    float d0 = dist[b * K2 + k] / nt;
    float d1 = dist[b * K2 + KP + k] / nt;
    float s0 = __expf(-0.5f * d0 - bb); // ALPHA = 0.5
    float s1 = __expf(-0.5f * d1 - bb);
    score0 += s0 * fmaxf(mlp_w[k], 0.f);
    score1 += s1 * fmaxf(mlp_w[KP + k], 0.f);
  }
  out[b] = score1 - score0;
}

extern "C" void kernel_launch(void* const* d_in, const int* in_sizes, int n_in,
                              void* d_out, int out_size, void* d_ws, size_t ws_size,
                              hipStream_t stream) {
  const float* sim   = (const float*)d_in[0];   // sim_itm (B,T,M)
  const float* ttl   = (const float*)d_in[1];   // test_tl (B,T)
  const int*   ntask = (const int*)d_in[2];     // n_task (B,)
  /* d_in[3] demo_bh unused */
  const float* basis = (const float*)d_in[4];   // (D,)
  const float* phase = (const float*)d_in[5];   // (D,)
  const float* W     = (const float*)d_in[6];   // (2K,M,2D)
  const float* mlpw  = (const float*)d_in[7];   // (2,K)

  char* ws = (char*)d_ws;
  _Float16* E  = (_Float16*)(ws + WS_E);
  _Float16* Wh = (_Float16*)(ws + WS_WH);
  float* bern  = (float*)(ws + WS_BERN);
  float* dist  = (float*)(ws + WS_DIST);
  float* out   = (float*)d_out;

  init_kernel<<<8, 256, 0, stream>>>(bern, dist);
  wconv_kernel<<<(K2 * M * 2 * DD) / 256, 256, 0, stream>>>(W, Wh);
  eprep_kernel<<<(B * T * 32) / 256, 256, 0, stream>>>(ttl, ntask, basis, phase, E);
  wmma_kernel<<<(B * 32) / 8, 256, 0, stream>>>(sim, E, Wh, bern, dist);
  final_kernel<<<1, 128, 0, stream>>>(ntask, mlpw, bern, dist, out);
}
